// BilinearAttention_50027779064479
// MI455X (gfx1250) — matrix operand order repaired and checked by reading, not compiled, not run on hardware
//
#include <hip/hip_runtime.h>
#include <hip/hip_bf16.h>
#include <math.h>

// Problem constants (match reference)
#define BB 128
#define PP 49
#define EE 2048
#define DD 512
#define AA 512

typedef __bf16 bf16_t;
typedef bf16_t bf16x16 __attribute__((ext_vector_type(16)));
typedef float  f32x8   __attribute__((ext_vector_type(8)));
typedef float  f32x4   __attribute__((ext_vector_type(4)));

union FragB16 {
    bf16x16 v;
    uint4   q[2];
    unsigned u[8];
};

// Pack two fp32 -> bf16 pair with a single v_perm_b32 (truncation).
// result[15:0] = lo[31:16], result[31:16] = hi[31:16]
__device__ __forceinline__ unsigned permpack(float lo, float hi) {
    return __builtin_amdgcn_perm(__builtin_bit_cast(unsigned, hi),
                                 __builtin_bit_cast(unsigned, lo), 0x07060302u);
}

// ---- A-matrix fragment (16x32 bf16): lane half h, lane-in-half lr.
// row = lr; elems 0..15 = K(h*16 .. h*16+15)
__device__ __forceinline__ bf16x16 load_a_f32(const float* src, int ld, int lane) {
    int h = lane >> 4, lr = lane & 15;
    const f32x4* p = (const f32x4*)(src + lr * ld + h * 16);
    f32x4 c0 = p[0], c1 = p[1];         // K chunk 1 (8 elems)
    f32x4 c2 = p[2], c3 = p[3];         // K chunk 2 (+8 floats = +2 f32x4)
    FragB16 f;
    f.u[0] = permpack(c0[0], c0[1]); f.u[1] = permpack(c0[2], c0[3]);
    f.u[2] = permpack(c1[0], c1[1]); f.u[3] = permpack(c1[2], c1[3]);
    f.u[4] = permpack(c2[0], c2[1]); f.u[5] = permpack(c2[2], c2[3]);
    f.u[6] = permpack(c3[0], c3[1]); f.u[7] = permpack(c3[2], c3[3]);
    return f.v;
}

// Non-temporal variant for single-pass streams (Wb): don't pollute L2.
__device__ __forceinline__ bf16x16 load_a_f32_nt(const float* src, int ld, int lane) {
    int h = lane >> 4, lr = lane & 15;
    const f32x4* p = (const f32x4*)(src + lr * ld + h * 16);
    f32x4 c0 = __builtin_nontemporal_load(p);
    f32x4 c1 = __builtin_nontemporal_load(p + 1);
    f32x4 c2 = __builtin_nontemporal_load(p + 2);
    f32x4 c3 = __builtin_nontemporal_load(p + 3);
    FragB16 f;
    f.u[0] = permpack(c0[0], c0[1]); f.u[1] = permpack(c0[2], c0[3]);
    f.u[2] = permpack(c1[0], c1[1]); f.u[3] = permpack(c1[2], c1[3]);
    f.u[4] = permpack(c2[0], c2[1]); f.u[5] = permpack(c2[2], c2[3]);
    f.u[6] = permpack(c3[0], c3[1]); f.u[7] = permpack(c3[2], c3[3]);
    return f.v;
}

__device__ __forceinline__ bf16x16 load_a_bf16_guard(const bf16_t* src, int ld, int lane, int rows) {
    int h = lane >> 4, lr = lane & 15;
    FragB16 f;
    if (lr < rows) {
        const bf16_t* p = src + lr * ld + h * 16;
        f.q[0] = *(const uint4*)p;
        f.q[1] = *(const uint4*)(p + 8);
    } else {
        f.q[0] = make_uint4(0u, 0u, 0u, 0u);
        f.q[1] = make_uint4(0u, 0u, 0u, 0u);
    }
    return f.v;
}

// ---- B-matrix fragment (32x16 bf16, column-major): lane = column,
// 16 contiguous K elements starting at K = h*16 (src is B^T row-major)
__device__ __forceinline__ bf16x16 load_b_f32(const float* src, int ld, int lane) {
    int h = lane >> 4, lr = lane & 15;
    const f32x4* p = (const f32x4*)(src + lr * ld + h * 16);
    f32x4 c0 = p[0], c1 = p[1], c2 = p[2], c3 = p[3];
    FragB16 f;
    f.u[0] = permpack(c0[0], c0[1]); f.u[1] = permpack(c0[2], c0[3]);
    f.u[2] = permpack(c1[0], c1[1]); f.u[3] = permpack(c1[2], c1[3]);
    f.u[4] = permpack(c2[0], c2[1]); f.u[5] = permpack(c2[2], c2[3]);
    f.u[6] = permpack(c3[0], c3[1]); f.u[7] = permpack(c3[2], c3[3]);
    return f.v;
}
__device__ __forceinline__ bf16x16 load_b_bf16(const bf16_t* src, int ld, int lane) {
    int h = lane >> 4, lr = lane & 15;
    const bf16_t* p = src + lr * ld + h * 16;
    FragB16 f;
    f.q[0] = *(const uint4*)p;
    f.q[1] = *(const uint4*)(p + 8);
    return f.v;
}

__device__ __forceinline__ f32x8 wmma_bf16(bf16x16 a, bf16x16 b, f32x8 c) {
    return __builtin_amdgcn_wmma_f32_16x16x32_bf16(false, a, false, b, (short)0, c, false, false);
}

// D fragment element (m,n): n = lane%16, m = 8*(lane/16) + g  (g = vector idx)
__device__ __forceinline__ void store_d_bf16(bf16_t* dst, f32x8 c) {
    uint4 q;
    q.x = permpack(c[0], c[1]);
    q.y = permpack(c[2], c[3]);
    q.z = permpack(c[4], c[5]);
    q.w = permpack(c[6], c[7]);
    *(uint4*)dst = q;
}

__device__ __forceinline__ float xor_reduce16(float v, int lane) {
#pragma unroll
    for (int mask = 1; mask <= 8; mask <<= 1) {
        int idx = ((lane ^ mask) << 2);
        v += __builtin_bit_cast(float, __builtin_amdgcn_ds_bpermute(idx, __builtin_bit_cast(int, v)));
    }
    return v;
}

// ============ K1: att2[b][a] = dh[b]·Wd[a] + bd[a]  (bf16 out) ============
// D[m=a, n=b] = Wd (A) x dh^T (B).  grid: 32 blocks x 256 thr (8 waves = 8 b-tiles)
__global__ void k1_att2(const float* __restrict__ dh, const float* __restrict__ Wd,
                        const float* __restrict__ bd, bf16_t* __restrict__ att2bf) {
    int lane = threadIdx.x & 31;
    int wid  = threadIdx.x >> 5;
    int a0 = blockIdx.x * 16;
    int b0 = wid * 16;
    int h = lane >> 4, lr = lane & 15;
    f32x8 acc = {};
    for (int kk = 0; kk < DD; kk += 32) {
        bf16x16 a = load_a_f32(Wd + a0 * DD + kk, DD, lane);
        bf16x16 b = load_b_f32(dh + b0 * DD + kk, DD, lane);
        acc = wmma_bf16(a, b, acc);
    }
#pragma unroll
    for (int g = 0; g < 8; ++g) acc[g] += bd[a0 + h * 8 + g];
    store_d_bf16(att2bf + (b0 + lr) * AA + a0 + h * 8, acc);
}

// ============ K2: att1[m=b*P+p][a] = enc[m]·We[a] + be[a] (bf16) ============
// D[m=a, n=flat].  grid: (32, 49) x 256 thr; wave -> ntile = by*8+wid
__global__ void k2_att1(const float* __restrict__ enc, const float* __restrict__ We,
                        const float* __restrict__ be, bf16_t* __restrict__ att1bf) {
    int lane = threadIdx.x & 31;
    int wid  = threadIdx.x >> 5;
    int a0 = blockIdx.x * 16;
    int n0 = (blockIdx.y * 8 + wid) * 16;   // flat (b*P+p) tile, 392 tiles exactly
    int h = lane >> 4, lr = lane & 15;
    f32x8 acc = {};
    for (int kk = 0; kk < EE; kk += 32) {
        bf16x16 a = load_a_f32(We + a0 * EE + kk, EE, lane);
        bf16x16 b = load_b_f32(enc + n0 * EE + kk, EE, lane);
        acc = wmma_bf16(a, b, acc);
    }
#pragma unroll
    for (int g = 0; g < 8; ++g) acc[g] += be[a0 + h * 8 + g];
    store_d_bf16(att1bf + (n0 + lr) * AA + a0 + h * 8, acc);
}

// ============ K3: U[b][k][i] = sum_j Wb[k][i][j] * att2[b][j]  (bf16) ============
// GEMM M=A*A rows of Wb, N=B, K=A. Single non-temporal pass over Wb (512 MB) — HBM bound.
// grid: 4096 blocks x 128 thr (4 waves, wave = 1 M-tile, 8 N-tiles in regs)
__global__ void k3_u(const float* __restrict__ Wb, const bf16_t* __restrict__ att2bf,
                     bf16_t* __restrict__ Ubf) {
    int lane = threadIdx.x & 31;
    int wid  = threadIdx.x >> 5;
    int m0 = (blockIdx.x * 4 + wid) * 16;   // row of Wb (k*A + i)
    int h = lane >> 4, lr = lane & 15;
    f32x8 acc[8];
#pragma unroll
    for (int nt = 0; nt < 8; ++nt) acc[nt] = f32x8{};
    for (int kk = 0; kk < AA; kk += 32) {
        bf16x16 a = load_a_f32_nt(Wb + (size_t)m0 * AA + kk, AA, lane);
#pragma unroll
        for (int nt = 0; nt < 8; ++nt) {
            bf16x16 b = load_b_bf16(att2bf + nt * 16 * AA + kk, AA, lane);
            acc[nt] = wmma_bf16(a, b, acc[nt]);
        }
    }
#pragma unroll
    for (int nt = 0; nt < 8; ++nt) {
        int bidx = nt * 16 + lr;
        store_d_bf16(Ubf + (size_t)bidx * (AA * AA) + m0 + h * 8, acc[nt]);
    }
}

// ============ K4: att[b][p] = relu(att1[b]·U[b]^T + bb)·Wf + bf ============
// Per b: M=p (4 tiles, padded), N=k (32 tiles), K=i (512).
// grid: 128 blocks x 128 thr (4 waves; wave = m-tile, loops all n-tiles).
// Each 16x512 U tile staged once per block in LDS via async global->LDS DMA.
__global__ void k4_bil(const bf16_t* __restrict__ att1bf, const bf16_t* __restrict__ Ubf,
                       const float* __restrict__ bb, const float* __restrict__ Wf,
                       const float* __restrict__ bfb, float* __restrict__ att) {
    __shared__ uint4 s_u[1024];            // 16 cols x 512 K of bf16 = 16 KB
    int tid  = threadIdx.x;
    int lane = tid & 31;
    int wid  = tid >> 5;
    int b = blockIdx.x;
    int p0 = wid * 16;
    int rows = PP - p0; if (rows > 16) rows = 16; if (rows < 0) rows = 0;
    int h = lane >> 4, lr = lane & 15;
    const bf16_t* a1 = att1bf + (size_t)b * PP * AA;
    const uint4* Ub4 = (const uint4*)(Ubf + (size_t)b * (AA * AA));
    float attp[8] = {0.f, 0.f, 0.f, 0.f, 0.f, 0.f, 0.f, 0.f};

    for (int nt = 0; nt < 32; ++nt) {
        __syncthreads();                    // previous tile fully consumed
        // Async DMA: 1024 uint4 (16 KB), 128 threads x 8 each, LDS dest per lane.
        const uint4* src = Ub4 + nt * 1024;
#pragma unroll
        for (int i = 0; i < 8; ++i) {
            int idx = i * 128 + tid;
            unsigned ldsoff = (unsigned)(size_t)(&s_u[idx]);
            unsigned long long ga = (unsigned long long)(const void*)(src + idx);
            asm volatile("global_load_async_to_lds_b128 %0, %1, off"
                         :: "v"(ldsoff), "v"(ga) : "memory");
        }
        asm volatile("s_wait_asynccnt 0x0" ::: "memory");
        __syncthreads();                    // tile visible to all waves

        f32x8 acc = {};
        for (int kk = 0; kk < AA; kk += 32) {
            bf16x16 a = load_a_bf16_guard(a1 + p0 * AA + kk, AA, lane, rows);
            // B fragment from LDS: column lr, K chunk h*16 within [kk, kk+32)
            int e = (lr * AA + kk + h * 16) >> 3;
            FragB16 bf;
            bf.q[0] = s_u[e];
            bf.q[1] = s_u[e + 1];
            acc = wmma_bf16(a, bf.v, acc);
        }
        int k = nt * 16 + lr;
        float bbv = bb[k];
        float wfv = Wf[k];
#pragma unroll
        for (int g = 0; g < 8; ++g) {
            float v = acc[g] + bbv;
            v = v > 0.f ? v : 0.f;
            attp[g] += v * wfv;
        }
    }
    float bf0 = bfb[0];
#pragma unroll
    for (int g = 0; g < 8; ++g) {
        float v = xor_reduce16(attp[g], lane);   // sum over the 16 n-lanes of this half
        if (lr == 0) {
            int p = p0 + h * 8 + g;
            if (p < PP) att[b * PP + p] = v + bf0;
        }
    }
}

// ============ K5: softmax over P + alpha-weighted encoder sum ============
__global__ void k5_softmax_awe(const float* __restrict__ att, const float* __restrict__ enc,
                               float* __restrict__ awe, float* __restrict__ alpha) {
    __shared__ float s[PP];
    __shared__ float s_inv;
    int b = blockIdx.x;
    int tid = threadIdx.x;
    if (tid < PP) s[tid] = att[b * PP + tid];
    __syncthreads();
    if (tid == 0) {
        float mx = -1e30f;
        for (int p = 0; p < PP; ++p) mx = fmaxf(mx, s[p]);
        float sum = 0.f;
        for (int p = 0; p < PP; ++p) { float e = __expf(s[p] - mx); s[p] = e; sum += e; }
        s_inv = 1.f / sum;
    }
    __syncthreads();
    if (tid < PP) {
        float a = s[tid] * s_inv;
        s[tid] = a;
        alpha[b * PP + tid] = a;
    }
    __syncthreads();
    const float* eb = enc + (size_t)b * PP * EE;
    for (int e = tid; e < EE; e += 256) {
        float acc = 0.f;
        for (int p = 0; p < PP; ++p) acc += eb[(size_t)p * EE + e] * s[p];
        awe[(size_t)b * EE + e] = acc;
    }
}

extern "C" void kernel_launch(void* const* d_in, const int* in_sizes, int n_in,
                              void* d_out, int out_size, void* d_ws, size_t ws_size,
                              hipStream_t stream) {
    const float* enc = (const float*)d_in[0];   // [B,P,E]
    const float* dh  = (const float*)d_in[1];   // [B,D]
    const float* We  = (const float*)d_in[2];   // [A,E]
    const float* be  = (const float*)d_in[3];   // [A]
    const float* Wd  = (const float*)d_in[4];   // [A,D]
    const float* bd  = (const float*)d_in[5];   // [A]
    const float* Wb  = (const float*)d_in[6];   // [A,A,A]
    const float* bb  = (const float*)d_in[7];   // [A]
    const float* Wf  = (const float*)d_in[8];   // [1,A]
    const float* bfb = (const float*)d_in[9];   // [1]

    float* out   = (float*)d_out;
    float* awe   = out;                          // [B,E]
    float* alpha = out + (size_t)BB * EE;        // [B,P]

    // workspace layout (bytes, 256-aligned)
    char* ws = (char*)d_ws;
    bf16_t* att2bf = (bf16_t*)(ws);                                  // 128*512*2      = 128 KB
    bf16_t* att1bf = (bf16_t*)(ws + 131072);                         // 6272*512*2     = 6.1 MB
    bf16_t* Ubf    = (bf16_t*)(ws + 6553600);                        // 128*512*512*2  = 64 MB
    float*  att    = (float*)(ws + 73662464);                        // 128*49*4

    k1_att2<<<dim3(AA / 16), dim3(256), 0, stream>>>(dh, Wd, bd, att2bf);
    k2_att1<<<dim3(AA / 16, (BB * PP) / (16 * 8)), dim3(256), 0, stream>>>(enc, We, be, att1bf);
    k3_u<<<dim3((AA * AA) / (16 * 4)), dim3(128), 0, stream>>>(Wb, att2bf, Ubf);
    k4_bil<<<dim3(BB), dim3(128), 0, stream>>>(att1bf, Ubf, bb, Wf, bfb, att);
    k5_softmax_awe<<<dim3(BB), dim3(256), 0, stream>>>(att, enc, awe, alpha);
}
